// AttentionOp_14216341750018
// MI455X (gfx1250) — matrix operand order repaired and checked by reading, not compiled, not run on hardware
//
#include <hip/hip_runtime.h>

// ---------------------------------------------------------------------------
// Chunked-causal GQA attention with sinks, flash-style, bf16 WMMA (gfx1250).
// q:[B,S,Hq,D] k,v:[B,S,Hkv,D] sinks:[Hq] -> out:[B,S,Hq,D]  (all fp32)
// - 256-thread blocks = 8 waves: waves 0-3 -> head 2*hkv, waves 4-7 -> head
//   2*hkv+1, sharing one K/V LDS tile (GQA-aware staging, halves KV traffic).
// - Softmax in log2 domain (scale*log2e folded into Q at staging).
// - Denominator (incl. sink term) accumulated via WMMA against a ones-matrix.
// - Row-max reduction via DPP (quad_perm / dpp8 / row_ror:8), not ds_bpermute.
// ---------------------------------------------------------------------------

#define S_LEN   2048
#define HQ      16
#define HKV     8
#define GQA     2
#define DH      128
#define CHUNK   1024
#define QTILE   64          // query rows per head per block (4 waves x 16)
#define KBLK    32          // keys per iteration (WMMA K for bf16)
#define NEG_BIG (-1.0e38f)
#define LOG2E   1.4426950408889634f
#define SCALE_LOG2E 0.1275274864670736f   // (1/sqrt(128)) * log2(e)

typedef __attribute__((ext_vector_type(16))) __bf16 v16bf;
typedef __attribute__((ext_vector_type(8)))  float  v8f;

union AFrag { v16bf v; unsigned int u[8]; };

__device__ __forceinline__ float fast_exp2(float x) {
    return __builtin_amdgcn_exp2f(x);                       // v_exp_f32
}
__device__ __forceinline__ unsigned short bf16bits(float f) {
    return __builtin_bit_cast(unsigned short, (__bf16)f);   // v_cvt_*_bf16
}
__device__ __forceinline__ unsigned int pack2bf(float a, float b) {
    return (unsigned int)bf16bits(a) | ((unsigned int)bf16bits(b) << 16);
}

// Cross-lane max over the 16-lane half, entirely on the VALU pipe via DPP.
// xor1: quad_perm [1,0,3,2]; xor2: quad_perm [2,3,0,1]; xor4: dpp8 lane^4;
// xor8: DPP16 row_ror:8 (rotate by 8 within a 16-lane row == lane^8).
template <int CTRL>
__device__ __forceinline__ float dpp_f32(float x) {
    int i = __builtin_bit_cast(int, x);
    int y = __builtin_amdgcn_update_dpp(i, i, CTRL, 0xF, 0xF, false);
    return __builtin_bit_cast(float, y);
}
__device__ __forceinline__ float dpp8_xor4(float x) {
    int i = __builtin_bit_cast(int, x);
    int y = __builtin_amdgcn_mov_dpp8(i, 0x688FAC);   // sel = [4,5,6,7,0,1,2,3]
    return __builtin_bit_cast(float, y);
}
__device__ __forceinline__ float rowmax16(float x) {
    x = fmaxf(x, dpp_f32<0xB1>(x));      // xor 1 (quad_perm [1,0,3,2])
    x = fmaxf(x, dpp_f32<0x4E>(x));      // xor 2 (quad_perm [2,3,0,1])
    x = fmaxf(x, dpp8_xor4(x));          // xor 4 (dpp8)
    return fmaxf(x, dpp_f32<0x128>(x));  // xor 8 (row_ror:8)
}

__global__ __launch_bounds__(256)
void attn_sink_chunked_kernel(const float* __restrict__ qp,
                              const float* __restrict__ kp,
                              const float* __restrict__ vp,
                              const float* __restrict__ sinks,
                              float* __restrict__ out)
{
    __shared__ alignas(16) unsigned short Kt[DH * KBLK];    // [d][key]   8 KB
    __shared__ alignas(16) unsigned short Vs[KBLK * DH];    // [key][d]   8 KB
    __shared__ alignas(16) unsigned short Ps[8][16 * KBLK]; // per-wave P, 8 KB
    __shared__ alignas(16) unsigned short Qs[8][16 * DH];   // per-wave Q, 32 KB

    const int tid   = threadIdx.x;
    const int wave  = tid >> 5;      // 0..7
    const int lane  = tid & 31;
    const int laneh = lane & 15;     // N / M position within 16
    const int hi    = lane >> 4;     // lane half

    const int tilesPerSeq = S_LEN / QTILE;               // 32
    int blk = blockIdx.x;
    int b   = blk / (HKV * tilesPerSeq);
    int rem = blk % (HKV * tilesPerSeq);
    int hkv = rem / tilesPerSeq;
    int qt  = rem % tilesPerSeq;
    int q0  = qt * QTILE;
    int g   = wave >> 2;                                 // GQA head within group
    int hq  = hkv * GQA + g;
    int chunk0 = (q0 / CHUNK) * CHUNK;
    int qw  = q0 + (wave & 3) * 16;                      // this wave's first q row

    const size_t qRow = (size_t)HQ * DH;
    const size_t kRow = (size_t)HKV * DH;
    const float* qBase = qp + (size_t)b * S_LEN * qRow + (size_t)hq  * DH;
    const float* kBase = kp + (size_t)b * S_LEN * kRow + (size_t)hkv * DH;
    const float* vBase = vp + (size_t)b * S_LEN * kRow + (size_t)hkv * DH;
    float*       oBase = out + (size_t)b * S_LEN * qRow + (size_t)hq * DH;

    // ---- stage this wave's 16x128 Q tile (pre-scaled by scale*log2e) ----
    #pragma unroll
    for (int i = 0; i < 16; ++i) {
        int lin = i * 32 + lane;                 // 0..511 float4s
        int r  = lin >> 5;                       // 0..15
        int d4 = (lin & 31) << 2;                // 0..124
        const float4 qq = *(const float4*)(qBase + (size_t)(qw + r) * qRow + d4);
        uint2 pq;
        pq.x = pack2bf(qq.x * SCALE_LOG2E, qq.y * SCALE_LOG2E);
        pq.y = pack2bf(qq.z * SCALE_LOG2E, qq.w * SCALE_LOG2E);
        *(uint2*)&Qs[wave][r * DH + d4] = pq;
    }
    asm volatile("s_wait_dscnt 0x0" ::: "memory");

    // A-fragment layout: lane -> M=laneh ; vgpr v -> K pair
    // k0 = (v>>2)*16 + hi*8 + (v&3)*2
    AFrag qf[4];
    #pragma unroll
    for (int f = 0; f < 4; ++f) {
        #pragma unroll
        for (int vv = 0; vv < 8; ++vv) {
            int k0 = (vv >> 2) * 16 + hi * 8 + (vv & 3) * 2 + 32 * f;
            qf[f].u[vv] = *(const unsigned int*)&Qs[wave][laneh * DH + k0];
        }
    }

    // ones B-matrix for WMMA row-sum (constant => layout-independent)
    AFrag onesf;
    #pragma unroll
    for (int j = 0; j < 8; ++j) onesf.u[j] = 0x3F803F80u;   // bf16 1.0 pair

    // ---- online-softmax state (per lane: 8 rows, m = r + 8*hi) ----
    float sink2 = sinks[hq] * LOG2E;
    float mrow[8];
    v8f acc[8];
    v8f dacc = {1.f,1.f,1.f,1.f,1.f,1.f,1.f,1.f};  // denominator; 1.0 seeds sink term
    #pragma unroll
    for (int r = 0; r < 8; ++r) mrow[r] = sink2;
    #pragma unroll
    for (int n = 0; n < 8; ++n) acc[n] = (v8f){};

    const int kend = q0 + QTILE;   // causal limit for the whole 64-row tile
    for (int kb = chunk0; kb < kend; kb += KBLK) {
        __syncthreads();           // protect Kt/Vs from previous iteration
        // cooperative staging of 32 keys x 128 d, float4 per thread (256 thr)
        #pragma unroll
        for (int i = 0; i < 4; ++i) {
            int lin = i * 256 + tid;             // 0..1023 float4s
            int key = lin >> 5;                  // 0..31
            int d4  = (lin & 31) << 2;           // 0..124
            const float4 kk = *(const float4*)(kBase + (size_t)(kb + key) * kRow + d4);
            Kt[(d4 + 0) * KBLK + key] = bf16bits(kk.x);   // transposed [d][key]
            Kt[(d4 + 1) * KBLK + key] = bf16bits(kk.y);
            Kt[(d4 + 2) * KBLK + key] = bf16bits(kk.z);
            Kt[(d4 + 3) * KBLK + key] = bf16bits(kk.w);
            const float4 vv = *(const float4*)(vBase + (size_t)(kb + key) * kRow + d4);
            uint2 pv;
            pv.x = pack2bf(vv.x, vv.y);
            pv.y = pack2bf(vv.z, vv.w);
            *(uint2*)&Vs[key * DH + d4] = pv;             // row-major [key][d]
        }
        __syncthreads();

        // ---- logits (log2 domain): two 16x16 tiles, 8 WMMAs ----
        v8f ct[2];
        #pragma unroll
        for (int t = 0; t < 2; ++t) {
            v8f c = (v8f){};
            #pragma unroll
            for (int f = 0; f < 4; ++f) {
                AFrag bk;   // B 32x16: lane = key column (laneh), elements over d (table k order)
                const unsigned short* src = &Kt[(32 * f) * KBLK + 16 * t + laneh];
                #pragma unroll
                for (int j = 0; j < 8; ++j) {
                    int k0 = (j >> 2) * 16 + hi * 8 + (j & 3) * 2;
                    bk.u[j] = (unsigned int)src[k0 * KBLK] | ((unsigned int)src[(k0 + 1) * KBLK] << 16);
                }
                c = __builtin_amdgcn_wmma_f32_16x16x32_bf16(
                        false, qf[f].v, false, bk.v, (short)0, c, false, false);
            }
            ct[t] = c;
        }

        // ---- online softmax (max via DPP; sums via WMMA below) ----
        const bool full = (kb + (KBLK - 1)) <= qw;   // wave-uniform: no mask needed
        #pragma unroll
        for (int r = 0; r < 8; ++r) {
            float l0 = ct[0][r];
            float l1 = ct[1][r];
            if (!full) {
                int qrow = qw + r + 8 * hi;
                if (kb + laneh      > qrow) l0 = NEG_BIG;
                if (kb + 16 + laneh > qrow) l1 = NEG_BIG;
            }
            float bm = rowmax16(fmaxf(l0, l1));
            float mnew = fmaxf(mrow[r], bm);
            float resc = fast_exp2(mrow[r] - mnew);
            mrow[r] = mnew;
            float p0 = fast_exp2(l0 - mnew);
            float p1 = fast_exp2(l1 - mnew);
            dacc[r] *= resc;                     // carries sink + old sums forward
            #pragma unroll
            for (int n = 0; n < 8; ++n) acc[n][r] *= resc;
            int m = r + 8 * hi;                  // C layout -> LDS P (bf16)
            Ps[wave][m * KBLK + laneh]      = bf16bits(p0);
            Ps[wave][m * KBLK + 16 + laneh] = bf16bits(p1);
        }
        asm volatile("s_wait_dscnt 0x0" ::: "memory");    // wave-private P RAW

        // ---- P (16x32) as A-fragment ----
        AFrag pf;
        #pragma unroll
        for (int vv = 0; vv < 8; ++vv) {
            int k0 = (vv >> 2) * 16 + hi * 8 + (vv & 3) * 2;
            pf.u[vv] = *(const unsigned int*)&Ps[wave][laneh * KBLK + k0];
        }

        // denominator row-sums: dacc += P x ones (1 WMMA, replaces reductions)
        dacc = __builtin_amdgcn_wmma_f32_16x16x32_bf16(
                   false, pf.v, false, onesf.v, (short)0, dacc, false, false);

        // ---- O += P x V : 8 WMMAs ----
        #pragma unroll
        for (int n = 0; n < 8; ++n) {
            AFrag vf;   // B 32x16: lane = d column (laneh), elements over keys (table k order)
            const unsigned short* src = &Vs[16 * n + laneh];
            #pragma unroll
            for (int j = 0; j < 8; ++j) {
                int k0 = (j >> 2) * 16 + hi * 8 + (j & 3) * 2;
                vf.u[j] = (unsigned int)src[k0 * DH] | ((unsigned int)src[(k0 + 1) * DH] << 16);
            }
            acc[n] = __builtin_amdgcn_wmma_f32_16x16x32_bf16(
                         false, pf.v, false, vf.v, (short)0, acc[n], false, false);
        }
    }

    // ---- normalize (denominator already includes sink term) and write ----
    #pragma unroll
    for (int r = 0; r < 8; ++r) {
        float inv = 1.0f / dacc[r];
        int qrow = qw + r + 8 * hi;
        float* orow = oBase + (size_t)qrow * qRow;
        #pragma unroll
        for (int n = 0; n < 8; ++n)
            orow[16 * n + laneh] = acc[n][r] * inv;
    }
}

extern "C" void kernel_launch(void* const* d_in, const int* in_sizes, int n_in,
                              void* d_out, int out_size, void* d_ws, size_t ws_size,
                              hipStream_t stream) {
    const float* q     = (const float*)d_in[0];
    const float* k     = (const float*)d_in[1];
    const float* v     = (const float*)d_in[2];
    const float* sinks = (const float*)d_in[3];
    float* out = (float*)d_out;
    int B = in_sizes[0] / (S_LEN * HQ * DH);                 // = 4
    dim3 grid((unsigned)(B * HKV * (S_LEN / QTILE)));        // 1024 blocks
    attn_sink_chunked_kernel<<<grid, 256, 0, stream>>>(q, k, v, sinks, out);
}
